// GodelLLM_30305289241251
// MI455X (gfx1250) — matrix operand order repaired and checked by reading, not compiled, not run on hardware
//
#include <hip/hip_runtime.h>

// ---------------- CDNA5 (gfx1250) wave32 WMMA types ----------------
typedef __attribute__((ext_vector_type(16))) __bf16 v16bf;
typedef __attribute__((ext_vector_type(8)))  float  v8f;

union FragU { v16bf v; uint4 q[2]; };

__device__ __forceinline__ v8f wmma_bf16(v16bf a, v16bf b, v8f c) {
  // D = A(16x32 bf16) x B(32x16 bf16) + C(16x16 f32)
  return __builtin_amdgcn_wmma_f32_16x16x32_bf16(false, a, false, b, (short)0, c, false, false);
}

__device__ __forceinline__ unsigned short f2bf(float f) {
  unsigned int u = __float_as_uint(f);
  u += 0x7FFFu + ((u >> 16) & 1u);          // round-to-nearest-even
  return (unsigned short)(u >> 16);
}

__device__ __forceinline__ float redmax16(float v) {
#pragma unroll
  for (int m = 1; m < 16; m <<= 1) v = fmaxf(v, __shfl_xor(v, m, 32));
  return v;
}
__device__ __forceinline__ float redsum16(float v) {
#pragma unroll
  for (int m = 1; m < 16; m <<= 1) v += __shfl_xor(v, m, 32);
  return v;
}

// ---------------- CDNA5 async global->LDS path (ASYNCcnt), guarded ----------------
#if defined(__has_builtin)
#  if __has_builtin(__builtin_amdgcn_global_load_async_to_lds_b128) && \
      __has_builtin(__builtin_amdgcn_s_wait_asynccnt)
#    define USE_ASYNC_LDS 1
#  endif
#endif
#ifndef USE_ASYNC_LDS
#  define USE_ASYNC_LDS 0
#endif

// Exact parameter types per clang diagnostic: int vector_size(16), AS1 src / AS3 dst
typedef int v4i __attribute__((vector_size(16)));
typedef __attribute__((address_space(1))) v4i v4i_g;
typedef __attribute__((address_space(3))) v4i v4i_l;

// copy 16B (8 bf16) global -> LDS
__device__ __forceinline__ void copy16_to_lds(const unsigned short* g, unsigned short* l) {
#if USE_ASYNC_LDS
  __builtin_amdgcn_global_load_async_to_lds_b128((v4i_g*)g, (v4i_l*)l, 0, 0);
#else
  *(uint4*)l = *(const uint4*)g;
#endif
}
__device__ __forceinline__ void async_join() {
#if USE_ASYNC_LDS
  __builtin_amdgcn_s_wait_asynccnt(0);
#endif
}

// ---------------- weight convert + transpose: f32[K][N] -> bf16[N][K] ----------------
__global__ __launch_bounds__(256) void cvt_tr_kernel(const float* __restrict__ src,
                                                     unsigned short* __restrict__ dst,
                                                     int K, int N) {
  int e = blockIdx.x * 256 + threadIdx.x;
  int k = e / N, n = e - k * N;
  dst[(size_t)n * K + k] = f2bf(src[e]);
}

// ---------------- per-token feature kernel ----------------
__global__ __launch_bounds__(256) void feat_kernel(const int* __restrict__ enc,
    const float* __restrict__ emb1_w, const float* __restrict__ emb1_b,
    const float* __restrict__ log_w,  const float* __restrict__ log_b,
    const float* __restrict__ prime_w,const float* __restrict__ prime_b,
    const float* __restrict__ div_w,  const float* __restrict__ div_b,
    const float* __restrict__ bit_w,  const float* __restrict__ bit_b,
    unsigned short* __restrict__ hbuf, unsigned short* __restrict__ feats) {
  const int t  = blockIdx.x;
  const int e  = enc[t];
  const float ef  = (float)e;
  const float nrm = ef * (1.0f / 2147483647.0f);
  const int tid = threadIdx.x;
#pragma unroll
  for (int j = tid; j < 512; j += 256) {
    float h = fmaxf(nrm * emb1_w[j] + emb1_b[j], 0.0f);
    hbuf[(size_t)t * 512 + j] = f2bf(h);
  }
  const int c = tid;
  float f0 = logf(ef + 1.0f) * log_w[c] + log_b[c];

  const int primes[10] = {2,3,5,7,11,13,17,19,23,29};
  float f1 = prime_b[c];
#pragma unroll
  for (int i = 0; i < 10; ++i)
    if (e % primes[i] == 0) f1 += prime_w[i * 256 + c];

  float f2 = div_b[c];
#pragma unroll
  for (int i = 0; i < 20; ++i) {
    int tv = i + 2;
    f2 += ((float)(e % tv) / (float)tv) * div_w[i * 256 + c];
  }

  float f3 = bit_b[c];
#pragma unroll
  for (int i = 0; i < 32; ++i)
    if ((e >> i) & 1) f3 += bit_w[i * 256 + c];

  size_t base = (size_t)t * 1024;
  feats[base +       c] = f2bf(f0);
  feats[base + 256 + c] = f2bf(f1);
  feats[base + 512 + c] = f2bf(f2);
  feats[base + 768 + c] = f2bf(f3);
}

// ---------------- generic bf16 WMMA GEMM (async double-buffered LDS) ----------------
// C[M][N] = A[M][K](bf16) * Wt[N][K](bf16, pre-transposed) + bias[N] (+C if ACCUM)
// OUTMODE: 0 = f32 only, 1 = f32 + bf16 mirror, 2 = bf16 only
template<int ACCUM, int OUTMODE>
__global__ __launch_bounds__(256) void gemm_bf16_kernel(
    const unsigned short* __restrict__ A,
    const unsigned short* __restrict__ Wt,
    const float* __restrict__ bias,
    float* __restrict__ Cf,
    unsigned short* __restrict__ Cb,
    int M, int N, int K) {
  __shared__ __align__(16) unsigned short As[2][128 * 32]; // [m][k]
  __shared__ __align__(16) unsigned short Bs[2][128 * 32]; // [n][k]
  const int tid   = threadIdx.x;
  const int warp  = tid >> 5, lane = tid & 31;
  const int lrow  = lane & 15, lhalf = lane >> 4;
  const int wm    = warp & 3,  wn    = warp >> 2;   // 4x2 wave grid
  const int mBase = blockIdx.y * 128, nBase = blockIdx.x * 128;
  const int srow  = tid >> 1, skoff = (tid & 1) * 16;

  v8f acc[2][4];
  const v8f vzero = {0,0,0,0,0,0,0,0};
#pragma unroll
  for (int i = 0; i < 2; ++i)
#pragma unroll
    for (int j = 0; j < 4; ++j) acc[i][j] = vzero;

  const unsigned short* gA = A  + (size_t)(mBase + srow) * K + skoff;
  const unsigned short* gB = Wt + (size_t)(nBase + srow) * K + skoff;

  auto stage = [&](int kk, int buf) {
    unsigned short* la = &As[buf][srow * 32 + skoff];
    unsigned short* lb = &Bs[buf][srow * 32 + skoff];
    copy16_to_lds(gA + kk,     la);
    copy16_to_lds(gA + kk + 8, la + 8);
    copy16_to_lds(gB + kk,     lb);
    copy16_to_lds(gB + kk + 8, lb + 8);
  };

  stage(0, 0);
  async_join();
  __syncthreads();

  int cur = 0;
  for (int kk = 0; kk < K; kk += 32) {
    if (kk + 32 < K) {
      stage(kk + 32, cur ^ 1);                     // fill next slab while computing
      if (kk + 64 < K) {                           // pull slab-after-next toward L2
        __builtin_prefetch(gA + kk + 64, 0, 1);
        __builtin_prefetch(gB + kk + 64, 0, 1);
      }
    }

    FragU a[2];
#pragma unroll
    for (int mt = 0; mt < 2; ++mt) {
      const unsigned short* p = &As[cur][(wm * 32 + mt * 16 + lrow) * 32];
      // 16-bit A layout: K(e) = (e>>3)*16 + lhalf*8 + (e&7)
      a[mt].q[0] = *(const uint4*)(p + lhalf * 8);
      a[mt].q[1] = *(const uint4*)(p + 16 + lhalf * 8);
    }
#pragma unroll
    for (int nt = 0; nt < 4; ++nt) {
      FragU b;
      const unsigned short* p = &Bs[cur][(wn * 64 + nt * 16 + lrow) * 32];
      // 16-bit B layout: K(e) = (e>>3)*16 + lhalf*8 + (e&7)
      b.q[0] = *(const uint4*)(p + lhalf * 8);
      b.q[1] = *(const uint4*)(p + 16 + lhalf * 8);
#pragma unroll
      for (int mt = 0; mt < 2; ++mt)
        acc[mt][nt] = wmma_bf16(a[mt].v, b.v, acc[mt][nt]);
    }

    async_join();
    __syncthreads();
    cur ^= 1;
  }

#pragma unroll
  for (int mt = 0; mt < 2; ++mt)
#pragma unroll
    for (int nt = 0; nt < 4; ++nt)
#pragma unroll
      for (int r = 0; r < 8; ++r) {
        int row = mBase + wm * 32 + mt * 16 + r + 8 * lhalf; // C layout: M = r + 8*half
        int col = nBase + wn * 64 + nt * 16 + lrow;          //          N = lane&15
        float v = acc[mt][nt][r] + bias[col];
        size_t idx = (size_t)row * N + col;
        if constexpr (ACCUM)        v += Cf[idx];
        if constexpr (OUTMODE != 2) Cf[idx] = v;
        if constexpr (OUTMODE >= 1) Cb[idx] = f2bf(v);
      }
}

// ---------------- flash attention with gcd bias ----------------
// grid: (S/64, B*H), block: 128 (4 waves). Wave w owns query rows [w*16, w*16+16).
__global__ __launch_bounds__(128) void flash_kernel(
    const unsigned short* __restrict__ Q,
    const unsigned short* __restrict__ Kb,
    const unsigned short* __restrict__ V,
    const int* __restrict__ enc,
    const float* __restrict__ arith_bias,
    const float* __restrict__ gcd_w,
    unsigned short* __restrict__ O) {
  const int S = 2048;
  const int b = blockIdx.y >> 4, h = blockIdx.y & 15;
  const int qBase = blockIdx.x * 64;
  const int tid = threadIdx.x;
  const int w = tid >> 5, lane = tid & 31;
  const int lrow = lane & 15, lhalf = lane >> 4;

  __shared__ __align__(16) unsigned short Qs[64 * 64]; // [i][d]
  __shared__ __align__(16) unsigned short Ks[64 * 64]; // [j][d]  (== B layout for Q K^T)
  __shared__ __align__(16) unsigned short Vt[64 * 64]; // [d][j]  (B layout for P V)
  __shared__ __align__(16) unsigned short Ps[64 * 64]; // [i][j]
  __shared__ int Ei[64];
  __shared__ int Ej[64];

#pragma unroll
  for (int it = 0; it < 4; ++it) {
    int idx = tid + it * 128;
    int row = idx >> 3, part = idx & 7;
    copy16_to_lds(Q + (size_t)(b * S + qBase + row) * 1024 + h * 64 + part * 8,
                  &Qs[row * 64 + part * 8]);
  }
  if (tid < 64) Ei[tid] = enc[b * S + qBase + tid];

  const float ab = arith_bias[h];
  const float gw = gcd_w[0];

  float rowM[8], rowL[8];
  v8f oac[4];
  const v8f vzero = {0,0,0,0,0,0,0,0};
#pragma unroll
  for (int r = 0; r < 8; ++r) { rowM[r] = -3.0e38f; rowL[r] = 0.0f; }
#pragma unroll
  for (int nt = 0; nt < 4; ++nt) oac[nt] = vzero;

  for (int kBase = 0; kBase < S; kBase += 64) {
    __syncthreads();           // previous chunk's compute done before restaging
#pragma unroll
    for (int it = 0; it < 4; ++it) {
      int idx = tid + it * 128;
      int row = idx >> 3, part = idx & 7;
      copy16_to_lds(Kb + (size_t)(b * S + kBase + row) * 1024 + h * 64 + part * 8,
                    &Ks[row * 64 + part * 8]);
    }
    { // stage V transposed: Vt[d][j]
      int j = tid >> 1, dbase = (tid & 1) * 32;
#pragma unroll
      for (int q4 = 0; q4 < 4; ++q4) {
        uint4 vv = *(const uint4*)(V + (size_t)(b * S + kBase + j) * 1024 + h * 64 + dbase + q4 * 8);
        const unsigned short* pv = (const unsigned short*)&vv;
#pragma unroll
        for (int m = 0; m < 8; ++m)
          Vt[(dbase + q4 * 8 + m) * 64 + j] = pv[m];
      }
    }
    if (tid < 64) Ej[tid] = enc[b * S + kBase + tid];
    async_join();              // covers Qs (first iter) + Ks asyncs
    __syncthreads();

    // ---- scores: 16x64 per wave via WMMA, K-dim = hd = 64 (2 steps of 32) ----
    v8f sac[4];
#pragma unroll
    for (int nt = 0; nt < 4; ++nt) sac[nt] = vzero;
#pragma unroll
    for (int kk2 = 0; kk2 < 2; ++kk2) {
      int kk = kk2 * 32;
      FragU qa;
      const unsigned short* qp = &Qs[(w * 16 + lrow) * 64 + kk];
      qa.q[0] = *(const uint4*)(qp + lhalf * 8);
      qa.q[1] = *(const uint4*)(qp + 16 + lhalf * 8);
#pragma unroll
      for (int nt = 0; nt < 4; ++nt) {
        FragU kf;
        const unsigned short* kp = &Ks[(nt * 16 + lrow) * 64 + kk];
        kf.q[0] = *(const uint4*)(kp + lhalf * 8);
        kf.q[1] = *(const uint4*)(kp + 16 + lhalf * 8);
        sac[nt] = wmma_bf16(qa.v, kf.v, sac[nt]);
      }
    }

    // ---- gcd/arith bias + online softmax ----
    float nmax[8];
#pragma unroll
    for (int r = 0; r < 8; ++r) nmax[r] = rowM[r];
#pragma unroll
    for (int nt = 0; nt < 4; ++nt) {
      int ej   = Ej[nt * 16 + lrow];
      int jcol = kBase + nt * 16 + lrow;
#pragma unroll
      for (int r = 0; r < 8; ++r) {
        int rloc = r + 8 * lhalf;
        int irow = qBase + w * 16 + rloc;
        int ei   = Ei[w * 16 + rloc];
        float s  = sac[nt][r] * 0.125f + ab;       // 1/sqrt(64)
        if (irow != jcol) {
          int e1 = ei % (ej + 1);
          int e2 = ej % (ei + 1);
          int mn = e1 < e2 ? e1 : e2;
          int mx = ei > ej ? ei : ej;
          s += gw * ((float)mn / (float)(mx + 1));
        }
        sac[nt][r] = s;
        nmax[r] = fmaxf(nmax[r], s);
      }
    }
#pragma unroll
    for (int r = 0; r < 8; ++r) nmax[r] = redmax16(nmax[r]);

    float alpha[8], csum[8];
#pragma unroll
    for (int r = 0; r < 8; ++r) { alpha[r] = __expf(rowM[r] - nmax[r]); csum[r] = 0.0f; }
#pragma unroll
    for (int nt = 0; nt < 4; ++nt)
#pragma unroll
      for (int r = 0; r < 8; ++r) {
        float p = __expf(sac[nt][r] - nmax[r]);
        sac[nt][r] = p;
        csum[r] += p;
      }
#pragma unroll
    for (int r = 0; r < 8; ++r) {
      csum[r] = redsum16(csum[r]);
      rowL[r] = rowL[r] * alpha[r] + csum[r];
      rowM[r] = nmax[r];
    }
#pragma unroll
    for (int nt = 0; nt < 4; ++nt)
#pragma unroll
      for (int r = 0; r < 8; ++r) oac[nt][r] *= alpha[r];

    // ---- write P (bf16) to wave-private LDS rows, then O += P V ----
#pragma unroll
    for (int nt = 0; nt < 4; ++nt)
#pragma unroll
      for (int r = 0; r < 8; ++r)
        Ps[(w * 16 + r + 8 * lhalf) * 64 + nt * 16 + lrow] = f2bf(sac[nt][r]);

#pragma unroll
    for (int kk2 = 0; kk2 < 2; ++kk2) {
      int kk = kk2 * 32;
      FragU pa;
      const unsigned short* pp = &Ps[(w * 16 + lrow) * 64 + kk];
      pa.q[0] = *(const uint4*)(pp + lhalf * 8);
      pa.q[1] = *(const uint4*)(pp + 16 + lhalf * 8);
#pragma unroll
      for (int nt = 0; nt < 4; ++nt) {
        FragU vf;
        const unsigned short* vp = &Vt[(nt * 16 + lrow) * 64 + kk];
        vf.q[0] = *(const uint4*)(vp + lhalf * 8);
        vf.q[1] = *(const uint4*)(vp + 16 + lhalf * 8);
        oac[nt] = wmma_bf16(pa.v, vf.v, oac[nt]);
      }
    }
  }

  // ---- epilogue: divide by row sum, write bf16 attention output ----
#pragma unroll
  for (int nt = 0; nt < 4; ++nt)
#pragma unroll
    for (int r = 0; r < 8; ++r) {
      int row = qBase + w * 16 + r + 8 * lhalf;
      int col = h * 64 + nt * 16 + lrow;
      float v = oac[nt][r] / rowL[r];
      O[(size_t)(b * S + row) * 1024 + col] = f2bf(v);
    }
}

// ---------------- host-side orchestration ----------------
extern "C" void kernel_launch(void* const* d_in, const int* in_sizes, int n_in,
                              void* d_out, int out_size, void* d_ws, size_t ws_size,
                              hipStream_t stream) {
  (void)in_sizes; (void)n_in; (void)out_size; (void)ws_size;
  const float* q_w     = (const float*)d_in[0];
  const float* q_b     = (const float*)d_in[1];
  const float* k_w     = (const float*)d_in[2];
  const float* k_b     = (const float*)d_in[3];
  const float* v_w     = (const float*)d_in[4];
  const float* v_b     = (const float*)d_in[5];
  const float* o_w     = (const float*)d_in[6];
  const float* o_b     = (const float*)d_in[7];
  const float* emb1_w  = (const float*)d_in[8];
  const float* emb1_b  = (const float*)d_in[9];
  const float* emb2_w  = (const float*)d_in[10];
  const float* emb2_b  = (const float*)d_in[11];
  const float* log_w   = (const float*)d_in[12];
  const float* log_b   = (const float*)d_in[13];
  const float* prime_w = (const float*)d_in[14];
  const float* prime_b = (const float*)d_in[15];
  const float* div_w   = (const float*)d_in[16];
  const float* div_b   = (const float*)d_in[17];
  const float* bit_w   = (const float*)d_in[18];
  const float* bit_b   = (const float*)d_in[19];
  const float* fus_w   = (const float*)d_in[20];
  const float* fus_b   = (const float*)d_in[21];
  const float* a_bias  = (const float*)d_in[22];
  const float* gcd_w   = (const float*)d_in[23];
  const int*   enc     = (const int*)d_in[24];

  const int T = 2 * 2048; // B*S = 4096 tokens

  char* ws = (char*)d_ws;
  size_t off = 0;
  auto take = [&](size_t bytes) -> char* {
    char* p = ws + off;
    off = (off + bytes + 255) & ~(size_t)255;
    return p;
  };
  unsigned short* hbuf    = (unsigned short*)take((size_t)T * 512 * 2);
  unsigned short* featsb  = (unsigned short*)take((size_t)T * 1024 * 2);
  float*          hiddenf = (float*)         take((size_t)T * 1024 * 4);
  unsigned short* hiddenb = (unsigned short*)take((size_t)T * 1024 * 2);
  unsigned short* qb      = (unsigned short*)take((size_t)T * 1024 * 2);
  unsigned short* kb      = (unsigned short*)take((size_t)T * 1024 * 2);
  unsigned short* vb      = (unsigned short*)take((size_t)T * 1024 * 2);
  unsigned short* attnb   = (unsigned short*)take((size_t)T * 1024 * 2);
  unsigned short* w_emb2  = (unsigned short*)take((size_t)512 * 1024 * 2);
  unsigned short* w_fus   = (unsigned short*)take((size_t)1024 * 1024 * 2);
  unsigned short* w_q     = (unsigned short*)take((size_t)1024 * 1024 * 2);
  unsigned short* w_k     = (unsigned short*)take((size_t)1024 * 1024 * 2);
  unsigned short* w_v     = (unsigned short*)take((size_t)1024 * 1024 * 2);
  unsigned short* w_o     = (unsigned short*)take((size_t)1024 * 1024 * 2);

  // 1) weights -> bf16, transposed to [N][K]
  cvt_tr_kernel<<<dim3((512 * 1024) / 256), 256, 0, stream>>>(emb2_w, w_emb2, 512, 1024);
  cvt_tr_kernel<<<dim3((1024 * 1024) / 256), 256, 0, stream>>>(fus_w, w_fus, 1024, 1024);
  cvt_tr_kernel<<<dim3((1024 * 1024) / 256), 256, 0, stream>>>(q_w, w_q, 1024, 1024);
  cvt_tr_kernel<<<dim3((1024 * 1024) / 256), 256, 0, stream>>>(k_w, w_k, 1024, 1024);
  cvt_tr_kernel<<<dim3((1024 * 1024) / 256), 256, 0, stream>>>(v_w, w_v, 1024, 1024);
  cvt_tr_kernel<<<dim3((1024 * 1024) / 256), 256, 0, stream>>>(o_w, w_o, 1024, 1024);

  // 2) per-token features
  feat_kernel<<<dim3(T), 256, 0, stream>>>(enc, emb1_w, emb1_b, log_w, log_b,
                                           prime_w, prime_b, div_w, div_b,
                                           bit_w, bit_b, hbuf, featsb);

  // 3) hidden = relu-mlp(emb) + feats @ fus_w  (second pass accumulates + bf16 mirror)
  gemm_bf16_kernel<0, 0><<<dim3(8, 32), 256, 0, stream>>>(hbuf, w_emb2, emb2_b, hiddenf, nullptr, T, 1024, 512);
  gemm_bf16_kernel<1, 1><<<dim3(8, 32), 256, 0, stream>>>(featsb, w_fus, fus_b, hiddenf, hiddenb, T, 1024, 1024);

  // 4) Q, K, V projections (bf16 outputs)
  gemm_bf16_kernel<0, 2><<<dim3(8, 32), 256, 0, stream>>>(hiddenb, w_q, q_b, nullptr, qb, T, 1024, 1024);
  gemm_bf16_kernel<0, 2><<<dim3(8, 32), 256, 0, stream>>>(hiddenb, w_k, k_b, nullptr, kb, T, 1024, 1024);
  gemm_bf16_kernel<0, 2><<<dim3(8, 32), 256, 0, stream>>>(hiddenb, w_v, v_b, nullptr, vb, T, 1024, 1024);

  // 5) flash attention with fused gcd/arith bias
  flash_kernel<<<dim3(2048 / 64, 2 * 16), 128, 0, stream>>>(qb, kb, vb, enc, a_bias, gcd_w, attnb);

  // 6) output projection -> d_out (f32)
  gemm_bf16_kernel<0, 0><<<dim3(8, 32), 256, 0, stream>>>(attnb, w_o, o_b, (float*)d_out, nullptr, T, 1024, 1024);
}
